// Attention_20246475833939
// MI455X (gfx1250) — matrix operand order repaired and checked by reading, not compiled, not run on hardware
//
#include <hip/hip_runtime.h>

// ---------------------------------------------------------------------------
// ViT attention block for MI455X (gfx1250, wave32, WMMA).
// Heavy math: v_wmma_f32_16x16x32_bf16 (fp32 accumulate).
// GEMM tiles move global->LDS via gfx1250 async-to-LDS (ASYNCcnt) when the
// toolchain exposes the builtin, with a VGPR-staged fallback otherwise.
// ---------------------------------------------------------------------------

#define Hh    16
#define Dh    64
#define Bsz   64
#define SEQ   197
#define Cdim  1024
#define M_TOT (Bsz * SEQ)      // 12608 rows
#define N_QKV 3072
#define SP    208              // q/k padded seq rows (13*16)
#define SPV   224              // v^T / P padded seq cols (7*32)
#define NT    13               // score tiles along key dim

typedef __attribute__((ext_vector_type(16))) __bf16        v16bf;
typedef __attribute__((ext_vector_type(8)))  float         v8f;
typedef __attribute__((ext_vector_type(16))) unsigned short us16;
typedef __attribute__((ext_vector_type(8)))  unsigned short us8;
typedef __attribute__((ext_vector_type(4)))  int           i32x4;

#define AS1 __attribute__((address_space(1)))
#define AS3 __attribute__((address_space(3)))

#if defined(__has_builtin)
#if __has_builtin(__builtin_amdgcn_global_load_async_to_lds_b128) && \
    __has_builtin(__builtin_amdgcn_s_wait_asynccnt)
#define USE_ASYNC_LDS 1
#endif
#endif
#ifndef USE_ASYNC_LDS
#define USE_ASYNC_LDS 0
#endif

__device__ __forceinline__ unsigned short f2bf(float f) {
  unsigned u = __float_as_uint(f);
  u += 0x7fffu + ((u >> 16) & 1u);     // round-to-nearest-even
  return (unsigned short)(u >> 16);
}

__device__ __forceinline__ v16bf mk16(us8 lo, us8 hi) {
  us16 t;
#pragma unroll
  for (int i = 0; i < 8; ++i) { t[i] = lo[i]; t[i + 8] = hi[i]; }
  return __builtin_bit_cast(v16bf, t);
}

#define WMMA_BF16(A, B, C) \
  __builtin_amdgcn_wmma_f32_16x16x32_bf16(false, (A), false, (B), (short)0, (C), false, false)

// 16-byte global -> LDS copy: async DMA path on gfx1250 toolchains that have
// it (no VGPR staging, tracked by ASYNCcnt), else staged copy.
__device__ __forceinline__ void cp_b128(const unsigned short* g, unsigned short* l) {
#if USE_ASYNC_LDS
  __builtin_amdgcn_global_load_async_to_lds_b128(
      (AS1 i32x4*)(uintptr_t)g, (AS3 i32x4*)(uintptr_t)l, 0, 0);
#else
  *(us8*)l = *(const us8*)g;
#endif
}

#if USE_ASYNC_LDS
#define WAIT_ASYNC(n) __builtin_amdgcn_s_wait_asynccnt(n)
#else
#define WAIT_ASYNC(n) ((void)0)
#endif

// ---------------------------------------------------------------------------
// fp32 -> bf16 conversion (grid-stride)
// ---------------------------------------------------------------------------
__global__ void cvt_bf16(const float* __restrict__ src,
                         unsigned short* __restrict__ dst, int n) {
  for (int i = blockIdx.x * blockDim.x + threadIdx.x; i < n;
       i += gridDim.x * blockDim.x)
    dst[i] = f2bf(src[i]);
}

// ---------------------------------------------------------------------------
// RoPE tables: cos/sin (196 x 64), tiled frequencies (concat([f,f]))
// ---------------------------------------------------------------------------
__global__ void rope_tables(float* __restrict__ cosT, float* __restrict__ sinT) {
  int i = blockIdx.x * blockDim.x + threadIdx.x;
  if (i >= 196 * 64) return;
  int s = i >> 6, d = i & 63;
  int j = d & 31;                                  // tiled: col d uses f[d%32]
  float base = powf(10000.f, -2.f * (float)j / 64.f);
  float f = (float)s * base;
  cosT[i] = cosf(f);
  sinT[i] = sinf(f);
}

// ---------------------------------------------------------------------------
// Zero the padded seq columns of v^T (cols 197..223) so P@V padding is clean.
// ---------------------------------------------------------------------------
__global__ void zero_vpad(unsigned short* __restrict__ vb) {
  int bh = blockIdx.x;  // 1024 = B*H
  for (int i = threadIdx.x; i < Dh * (SPV - SEQ); i += blockDim.x) {
    int d = i / (SPV - SEQ);
    int t = SEQ + i % (SPV - SEQ);
    vb[(size_t)(bh * Dh + d) * SPV + t] = 0;
  }
}

// ---------------------------------------------------------------------------
// Tiled WMMA GEMM core: C[128x64] per block, 8 waves of 32x32, BK=32,
// double-buffered LDS with async global->LDS tile transfers.
// A: [M][1024] bf16 (K-major)   B: [Nout][1024] bf16 (K-major = W row-major)
// ---------------------------------------------------------------------------
#define BM 128
#define BN 64
#define BK 32
#define LP 40   // LDS row pitch in elements (padding kills bank conflicts)

// Issue the 3 per-thread 16B transfers for one (A,B) K-tile.
__device__ __forceinline__ void tile_issue(
    const unsigned short* __restrict__ A, const unsigned short* __restrict__ Bw,
    unsigned short* AsBuf, unsigned short* BsBuf,
    int tM, int tN, int k0, int tid)
{
  const int ar = tid >> 1, ac = (tid & 1) * 16;   // A: 128x32, 16 elems/thread
  const int br = tid >> 2, bc = (tid & 3) * 8;    // B: 64x32,   8 elems/thread
  int gr = tM + ar;
  if (gr >= M_TOT) gr = M_TOT - 1;   // clamp: rows >= M_TOT are never stored
  const unsigned short* pa = A + (size_t)gr * Cdim + k0 + ac;
  cp_b128(pa,     &AsBuf[ar * LP + ac]);
  cp_b128(pa + 8, &AsBuf[ar * LP + ac + 8]);
  cp_b128(Bw + (size_t)(tN + br) * Cdim + k0 + bc, &BsBuf[br * LP + bc]);
}

// One BK=32 step: 2x2 fragments, 4 WMMA.
__device__ __forceinline__ void tile_mma(
    const unsigned short* AsBuf, const unsigned short* BsBuf,
    int wM, int wN, int ln, int half, v8f acc[2][2])
{
  v16bf av[2], bv[2];
#pragma unroll
  for (int ti = 0; ti < 2; ++ti) {
    int row = wM + ti * 16 + ln;
    av[ti] = mk16(*(const us8*)&AsBuf[row * LP + 8 * half],
                  *(const us8*)&AsBuf[row * LP + 16 + 8 * half]);
  }
#pragma unroll
  for (int tj = 0; tj < 2; ++tj) {
    int col = wN + tj * 16 + ln;
    bv[tj] = mk16(*(const us8*)&BsBuf[col * LP + 8 * half],
                  *(const us8*)&BsBuf[col * LP + 16 + 8 * half]);
  }
#pragma unroll
  for (int ti = 0; ti < 2; ++ti)
#pragma unroll
    for (int tj = 0; tj < 2; ++tj)
      acc[ti][tj] = WMMA_BF16(av[ti], bv[tj], acc[ti][tj]);
}

// Shared main loop body for both GEMMs.
#define GEMM_MAIN_LOOP(Aptr, Bptr)                                            \
  tile_issue(Aptr, Bptr, As[0], Bs[0], tM, tN, 0, tid);                       \
  int cur = 0;                                                                \
  for (int k0 = 0; k0 < Cdim - BK; k0 += BK) {                                \
    tile_issue(Aptr, Bptr, As[cur ^ 1], Bs[cur ^ 1], tM, tN, k0 + BK, tid);   \
    WAIT_ASYNC(3);            /* drain tile-k loads, keep k+1 in flight */    \
    __syncthreads();                                                          \
    tile_mma(As[cur], Bs[cur], wM, wN, ln, half, acc);                        \
    __syncthreads();          /* done reading before k+2 overwrites */        \
    cur ^= 1;                                                                 \
  }                                                                           \
  WAIT_ASYNC(0);                                                              \
  __syncthreads();                                                            \
  tile_mma(As[cur], Bs[cur], wM, wN, ln, half, acc);

// QKV GEMM with bias + RoPE + q/k/v scatter epilogue
__global__ __launch_bounds__(256) void qkv_gemm(
    const unsigned short* __restrict__ A,     // x_bf16  [12608][1024]
    const unsigned short* __restrict__ Bw,    // qkv_w   [3072][1024]
    const float* __restrict__ bias,           // [3072]
    const float* __restrict__ cosT, const float* __restrict__ sinT,
    unsigned short* __restrict__ qb,          // [B*H][208][64]
    unsigned short* __restrict__ kb,          // [B*H][208][64]
    unsigned short* __restrict__ vb)          // [B*H][64][224]  (v^T)
{
  __shared__ __align__(16) unsigned short As[2][BM * LP];
  __shared__ __align__(16) unsigned short Bs[2][BN * LP];
  const int tid  = threadIdx.x;
  const int wave = tid >> 5, lane = tid & 31;
  const int half = lane >> 4, ln = lane & 15;
  const int tM = blockIdx.x * BM, tN = blockIdx.y * BN;
  const int wM = (wave & 3) * 32, wN = (wave >> 2) * 32;

  v8f acc[2][2] = {};
  GEMM_MAIN_LOOP(A, Bw);

  // ---- epilogue: bias, RoPE (pair partner = adjacent lane), scatter -------
#pragma unroll
  for (int ti = 0; ti < 2; ++ti) {
#pragma unroll
    for (int tj = 0; tj < 2; ++tj) {
      int n_glob = tN + wN + tj * 16 + ln;
      float bval = bias[n_glob];
      int which = n_glob >> 10;            // 0=q 1=k 2=v
      int c = n_glob & 1023, h = c >> 6, d = c & 63;
#pragma unroll
      for (int r = 0; r < 8; ++r) {
        int m = tM + wM + ti * 16 + (half << 3) + r;
        float val = acc[ti][tj][r] + bval;
        float partner = __shfl_xor(val, 1, 32);  // uniform: all lanes execute
        if (m < M_TOT) {
          int b = m / SEQ, tok = m - b * SEQ;
          int bh = b * Hh + h;
          if (which == 2) {
            vb[((size_t)bh * Dh + d) * SPV + tok] = f2bf(val);
          } else {
            float o = val;
            if (tok >= 1) {
              int s = tok - 1;
              float cs = cosT[s * 64 + d], sn = sinT[s * 64 + d];
              o = (d & 1) ? (val * cs + partner * sn)
                          : (val * cs - partner * sn);
            }
            unsigned short* dst = (which == 0) ? qb : kb;
            dst[((size_t)bh * SP + tok) * Dh + d] = f2bf(o);
          }
        }
      }
    }
  }
}

// ---------------------------------------------------------------------------
// Fused attention per (b,h): scores -> masked softmax -> P@V.
// One workgroup per (b,h); each wave owns 16 q rows; P staged in per-wave LDS.
// ---------------------------------------------------------------------------
__global__ __launch_bounds__(256) void attn_kernel(
    const unsigned short* __restrict__ qb,
    const unsigned short* __restrict__ kb,
    const unsigned short* __restrict__ vb,
    unsigned short* __restrict__ ao)       // [12608][1024] bf16
{
  __shared__ __align__(16) unsigned short P[8 * 16 * SPV];  // 57344 B
  const int bh = blockIdx.x;
  const int b = bh >> 4, h = bh & 15;
  const int wave = threadIdx.x >> 5, lane = threadIdx.x & 31;
  const int half = lane >> 4, ln = lane & 15;
  const unsigned short* qp = qb + (size_t)bh * SP * Dh;
  const unsigned short* kp = kb + (size_t)bh * SP * Dh;
  const unsigned short* vp = vb + (size_t)bh * Dh * SPV;
  unsigned short* Pw = &P[wave * 16 * SPV];

  for (int qt = wave; qt < NT; qt += 8) {
    // A fragments: q rows qt*16..+15, K = d in {0..31},{32..63}
    const unsigned short* qr = qp + (size_t)(qt * 16 + ln) * Dh;
    v16bf aq0 = mk16(*(const us8*)(qr + 8 * half),
                     *(const us8*)(qr + 16 + 8 * half));
    v16bf aq1 = mk16(*(const us8*)(qr + 32 + 8 * half),
                     *(const us8*)(qr + 48 + 8 * half));

    // ---- scores: 13 tiles x (2 WMMA over d) --------------------------------
    v8f sc[NT];
#pragma unroll
    for (int kt = 0; kt < NT; ++kt) {
      const unsigned short* kr = kp + (size_t)(kt * 16 + ln) * Dh;
      v16bf bk0 = mk16(*(const us8*)(kr + 8 * half),
                       *(const us8*)(kr + 16 + 8 * half));
      v16bf bk1 = mk16(*(const us8*)(kr + 32 + 8 * half),
                       *(const us8*)(kr + 48 + 8 * half));
      v8f s = {};
      s = WMMA_BF16(aq0, bk0, s);
      s = WMMA_BF16(aq1, bk1, s);
      sc[kt] = s;
    }

    // ---- masked softmax (row m lives across lanes 0..15 of this half) -----
    float lrow[8];
#pragma unroll
    for (int r = 0; r < 8; ++r) {
      float mx = -INFINITY;
#pragma unroll
      for (int kt = 0; kt < NT; ++kt) {
        float v = sc[kt][r] * 0.125f;            // scale = D^-0.5
        int col = kt * 16 + ln;
        v = (col < SEQ) ? v : -INFINITY;         // mask padded keys
        sc[kt][r] = v;
        mx = fmaxf(mx, v);
      }
      mx = fmaxf(mx, __shfl_xor(mx, 1, 32));
      mx = fmaxf(mx, __shfl_xor(mx, 2, 32));
      mx = fmaxf(mx, __shfl_xor(mx, 4, 32));
      mx = fmaxf(mx, __shfl_xor(mx, 8, 32));
      float sum = 0.f;
#pragma unroll
      for (int kt = 0; kt < NT; ++kt) {
        float e = __expf(sc[kt][r] - mx);
        sc[kt][r] = e;
        sum += e;
      }
      sum += __shfl_xor(sum, 1, 32);
      sum += __shfl_xor(sum, 2, 32);
      sum += __shfl_xor(sum, 4, 32);
      sum += __shfl_xor(sum, 8, 32);
      lrow[r] = sum;
    }

    // ---- stage P (bf16) to per-wave LDS slab -------------------------------
#pragma unroll
    for (int kt = 0; kt < NT; ++kt)
#pragma unroll
      for (int r = 0; r < 8; ++r)
        Pw[((half << 3) + r) * SPV + kt * 16 + ln] = f2bf(sc[kt][r]);
#pragma unroll
    for (int r = 0; r < 8; ++r)                    // zero K-pad cols 208..223
      Pw[((half << 3) + r) * SPV + 208 + ln] = 0;
    // wave-local LDS producer->consumer: wait for our own DS stores
    asm volatile("s_wait_dscnt 0" ::: "memory");

    // ---- out = P @ V : 7 K-chunks x 4 d-tiles ------------------------------
    v8f oacc[4] = {};
#pragma unroll
    for (int kc = 0; kc < 7; ++kc) {
      int k0 = kc * 32;
      const unsigned short* pr = &Pw[ln * SPV + k0];
      v16bf ap = mk16(*(const us8*)(pr + 8 * half),
                      *(const us8*)(pr + 16 + 8 * half));
#pragma unroll
      for (int dt = 0; dt < 4; ++dt) {
        const unsigned short* vr = vp + (size_t)(dt * 16 + ln) * SPV + k0;
        v16bf bvv = mk16(*(const us8*)(vr + 8 * half),
                         *(const us8*)(vr + 16 + 8 * half));
        oacc[dt] = WMMA_BF16(ap, bvv, oacc[dt]);
      }
    }

    // ---- normalize + store (B,N,H*D) bf16 ----------------------------------
#pragma unroll
    for (int dt = 0; dt < 4; ++dt) {
      int dcol = dt * 16 + ln;
#pragma unroll
      for (int r = 0; r < 8; ++r) {
        int tok = qt * 16 + (half << 3) + r;
        if (tok < SEQ)
          ao[((size_t)(b * SEQ + tok)) * Cdim + h * Dh + dcol] =
              f2bf(oacc[dt][r] / lrow[r]);
      }
    }
  }
}

// ---------------------------------------------------------------------------
// Projection GEMM: out = attn_out @ proj_w^T + proj_b  (fp32 output)
// ---------------------------------------------------------------------------
__global__ __launch_bounds__(256) void proj_gemm(
    const unsigned short* __restrict__ A,     // attn_out bf16 [12608][1024]
    const unsigned short* __restrict__ Bw,    // proj_w bf16 [1024][1024]
    const float* __restrict__ bias,           // [1024]
    float* __restrict__ out)                  // [12608][1024] fp32
{
  __shared__ __align__(16) unsigned short As[2][BM * LP];
  __shared__ __align__(16) unsigned short Bs[2][BN * LP];
  const int tid  = threadIdx.x;
  const int wave = tid >> 5, lane = tid & 31;
  const int half = lane >> 4, ln = lane & 15;
  const int tM = blockIdx.x * BM, tN = blockIdx.y * BN;
  const int wM = (wave & 3) * 32, wN = (wave >> 2) * 32;

  v8f acc[2][2] = {};
  GEMM_MAIN_LOOP(A, Bw);

#pragma unroll
  for (int ti = 0; ti < 2; ++ti)
#pragma unroll
    for (int tj = 0; tj < 2; ++tj) {
      int n_glob = tN + wN + tj * 16 + ln;
      float bval = bias[n_glob];
#pragma unroll
      for (int r = 0; r < 8; ++r) {
        int m = tM + wM + ti * 16 + (half << 3) + r;
        if (m < M_TOT)
          out[(size_t)m * Cdim + n_glob] = acc[ti][tj][r] + bval;
      }
    }
}

// ---------------------------------------------------------------------------
// Workspace layout (bytes). Total ~144 MB.
// ---------------------------------------------------------------------------
#define OFF_XBF   ((size_t)0)
#define OFF_WQKV  (OFF_XBF  + (size_t)M_TOT * Cdim * 2)
#define OFF_WPROJ (OFF_WQKV + (size_t)N_QKV * Cdim * 2)
#define OFF_COS   (OFF_WPROJ + (size_t)Cdim * Cdim * 2)
#define OFF_SIN   (OFF_COS  + (size_t)196 * 64 * 4)
#define OFF_Q     (OFF_SIN  + (size_t)196 * 64 * 4)
#define OFF_K     (OFF_Q    + (size_t)Bsz * Hh * SP * Dh * 2)
#define OFF_V     (OFF_K    + (size_t)Bsz * Hh * SP * Dh * 2)
#define OFF_AO    (OFF_V    + (size_t)Bsz * Hh * Dh * SPV * 2)

extern "C" void kernel_launch(void* const* d_in, const int* in_sizes, int n_in,
                              void* d_out, int out_size, void* d_ws,
                              size_t ws_size, hipStream_t stream) {
  const float* x      = (const float*)d_in[0];
  const float* qkv_w  = (const float*)d_in[1];
  const float* qkv_b  = (const float*)d_in[2];
  const float* proj_w = (const float*)d_in[3];
  const float* proj_b = (const float*)d_in[4];

  char* ws = (char*)d_ws;
  unsigned short* xbf    = (unsigned short*)(ws + OFF_XBF);
  unsigned short* wqkv   = (unsigned short*)(ws + OFF_WQKV);
  unsigned short* wproj  = (unsigned short*)(ws + OFF_WPROJ);
  float*          cosT   = (float*)(ws + OFF_COS);
  float*          sinT   = (float*)(ws + OFF_SIN);
  unsigned short* qbuf   = (unsigned short*)(ws + OFF_Q);
  unsigned short* kbuf   = (unsigned short*)(ws + OFF_K);
  unsigned short* vbuf   = (unsigned short*)(ws + OFF_V);
  unsigned short* aobf   = (unsigned short*)(ws + OFF_AO);

  cvt_bf16<<<4096, 256, 0, stream>>>(x, xbf, M_TOT * Cdim);
  cvt_bf16<<<2048, 256, 0, stream>>>(qkv_w, wqkv, N_QKV * Cdim);
  cvt_bf16<<<1024, 256, 0, stream>>>(proj_w, wproj, Cdim * Cdim);
  rope_tables<<<(196 * 64 + 255) / 256, 256, 0, stream>>>(cosT, sinT);
  zero_vpad<<<Bsz * Hh, 256, 0, stream>>>(vbuf);

  dim3 gq((M_TOT + BM - 1) / BM, N_QKV / BN);   // 99 x 48
  qkv_gemm<<<gq, 256, 0, stream>>>(xbf, wqkv, qkv_b, cosT, sinT,
                                   qbuf, kbuf, vbuf);

  attn_kernel<<<Bsz * Hh, 256, 0, stream>>>(qbuf, kbuf, vbuf, aobf);

  dim3 gp((M_TOT + BM - 1) / BM, Cdim / BN);    // 99 x 16
  proj_gemm<<<gp, 256, 0, stream>>>(aobf, wproj, proj_b, (float*)d_out);
}
